// CausalSelfAttention_4037269258617
// MI455X (gfx1250) — matrix operand order repaired and checked by reading, not compiled, not run on hardware
//
#include <hip/hip_runtime.h>

typedef __attribute__((ext_vector_type(16))) _Float16 v16h;
typedef __attribute__((ext_vector_type(8)))  _Float16 v8h;
typedef __attribute__((ext_vector_type(4)))  _Float16 v4h;
typedef __attribute__((ext_vector_type(8)))  float    v8f;

#define EMBED 1024
#define NHEAD 16
#define HDIM  64
#define BATCH 4
#define SEQ   2048
#define MTOT  (BATCH * SEQ)

// ---------------------------------------------------------------------------
// Fragment gathers from LDS, per CDNA5 ISA VGPR layouts.
// Both loaders are two aligned 16-byte LDS reads per lane (ds_load_b128).
// ---------------------------------------------------------------------------
__device__ inline v16h frag_a(const _Float16* base, int stride, int lane) {
  const int r = lane & 15, g = lane >> 4;
  const _Float16* p = base + r * stride + g * 8;
  const v8h lo = *(const v8h*)p;          // K = g*8   .. g*8+7
  const v8h hi = *(const v8h*)(p + 16);   // K = 16+g*8 .. 16+g*8+7
  return __builtin_shufflevector(lo, hi, 0, 1, 2, 3, 4, 5, 6, 7,
                                 8, 9, 10, 11, 12, 13, 14, 15);
}

// B (32x16 f16) from a K-major source [n][k] (i.e. B^T row-major).
__device__ inline v16h frag_b_kmaj(const _Float16* base, int stride, int lane) {
  const int n = lane & 15, kb = (lane >> 4) * 8;
  const _Float16* p = base + n * stride + kb;
  const v8h lo = *(const v8h*)p;          // K = kb .. kb+7
  const v8h hi = *(const v8h*)(p + 16);   // K = 16+kb .. 16+kb+7
  return __builtin_shufflevector(lo, hi, 0, 1, 2, 3, 4, 5, 6, 7,
                                 8, 9, 10, 11, 12, 13, 14, 15);
}

// ---------------------------------------------------------------------------
// Tiled WMMA GEMM:  out = (A[MxK] * W[KxN] + bias) * scale
// Block 256 threads (8 wave32), tile 128x128, K-step 64 -> 16 WMMA per barrier.
// Software-pipelined: next K-tile is loaded into REGISTERS while the current
// tile's WMMAs execute (global latency hidden under matrix work).
// W tile staged TRANSPOSED (K-major) so B fragments are vector LDS loads.
// PERMUTE=true scatters output into [B, H, S, D] (f16) for attention.
// ---------------------------------------------------------------------------
template <typename AT, typename OT, bool PERMUTE>
__global__ __launch_bounds__(256) void gemm_wmma(
    const AT* __restrict__ A, const float* __restrict__ W,
    const float* __restrict__ bias, OT* __restrict__ out,
    int M, int N, int K, float scale) {
  constexpr int BM = 128, BN = 128, BK = 64;
  constexpr int LDA = BK + 8;  // 72 halves: 16B-aligned rows, conflict-free
  constexpr int LDB = BK + 8;  // BsT is [n][k], stride 72
  __shared__ __align__(16) _Float16 As[BM * LDA];
  __shared__ __align__(16) _Float16 BsT[BN * LDB];

  const int tid  = threadIdx.x;
  const int lane = tid & 31;
  const int wave = tid >> 5;
  const int wm   = wave >> 1;  // 0..3 : 32-row strip
  const int wn   = wave & 1;   // 0..1 : 64-col strip
  const int m0   = blockIdx.y * BM;
  const int n0   = blockIdx.x * BN;

  v8f acc[2][4] = {};

  // register staging buffers for the incoming tile
  float4 aRegF[8];
  v8h    aRegH[4];
  float4 bReg[8];

  auto load_tile = [&](int k0) {
    if constexpr (sizeof(AT) == 4) {
#pragma unroll
      for (int i = 0; i < 8; ++i) {
        const int idx = tid + i * 256;  // 2048 float4 = 128x64
        const int r = idx >> 4, c = (idx & 15) * 4;
        aRegF[i] = *(const float4*)&A[(size_t)(m0 + r) * K + k0 + c];
      }
    } else {
#pragma unroll
      for (int i = 0; i < 4; ++i) {
        const int idx = tid + i * 256;  // 1024 v8h = 128x64
        const int r = idx >> 3, c = (idx & 7) * 8;
        aRegH[i] = *(const v8h*)&A[(size_t)(m0 + r) * K + k0 + c];
      }
    }
#pragma unroll
    for (int i = 0; i < 8; ++i) {
      const int idx = tid + i * 256;    // 2048 float4 = 64x128
      const int k = idx >> 5, n = (idx & 31) * 4;
      bReg[i] = *(const float4*)&W[(size_t)(k0 + k) * N + n0 + n];
    }
  };

  auto store_tile = [&]() {
    if constexpr (sizeof(AT) == 4) {
#pragma unroll
      for (int i = 0; i < 8; ++i) {
        const int idx = tid + i * 256;
        const int r = idx >> 4, c = (idx & 15) * 4;
        v4h h;
        h[0] = (_Float16)aRegF[i].x; h[1] = (_Float16)aRegF[i].y;
        h[2] = (_Float16)aRegF[i].z; h[3] = (_Float16)aRegF[i].w;
        *(v4h*)&As[r * LDA + c] = h;
      }
    } else {
#pragma unroll
      for (int i = 0; i < 4; ++i) {
        const int idx = tid + i * 256;
        const int r = idx >> 3, c = (idx & 7) * 8;
        *(v8h*)&As[r * LDA + c] = aRegH[i];
      }
    }
#pragma unroll
    for (int i = 0; i < 8; ++i) {
      const int idx = tid + i * 256;
      const int k = idx >> 5, n = (idx & 31) * 4;
      BsT[(n + 0) * LDB + k] = (_Float16)bReg[i].x;
      BsT[(n + 1) * LDB + k] = (_Float16)bReg[i].y;
      BsT[(n + 2) * LDB + k] = (_Float16)bReg[i].z;
      BsT[(n + 3) * LDB + k] = (_Float16)bReg[i].w;
    }
  };

  load_tile(0);
  for (int k0 = 0; k0 < K; k0 += BK) {
    store_tile();
    __syncthreads();
    if (k0 + BK < K) load_tile(k0 + BK);  // overlap with WMMAs below

#pragma unroll
    for (int kc = 0; kc < 2; ++kc) {
      v16h a[2], b[4];
#pragma unroll
      for (int i = 0; i < 2; ++i)
        a[i] = frag_a(&As[(wm * 32 + i * 16) * LDA + kc * 32], LDA, lane);
#pragma unroll
      for (int j = 0; j < 4; ++j)
        b[j] = frag_b_kmaj(&BsT[(wn * 64 + j * 16) * LDB + kc * 32], LDB, lane);
#pragma unroll
      for (int i = 0; i < 2; ++i)
#pragma unroll
        for (int j = 0; j < 4; ++j)
          acc[i][j] = __builtin_amdgcn_wmma_f32_16x16x32_f16(
              false, a[i], false, b[j], (short)0, acc[i][j], false, false);
    }
    __syncthreads();
  }

  // ---- epilogue: C layout lanes 0-15 N=lane row v / lanes 16-31 row 8+v ----
  const int cn = lane & 15, g = lane >> 4;
#pragma unroll
  for (int i = 0; i < 2; ++i)
#pragma unroll
    for (int j = 0; j < 4; ++j)
#pragma unroll
      for (int v = 0; v < 8; ++v) {
        const int row = m0 + wm * 32 + i * 16 + g * 8 + v;
        const int col = n0 + wn * 64 + j * 16 + cn;
        const float val = (acc[i][j][v] + bias[col]) * scale;
        if constexpr (PERMUTE) {
          const int bb = row >> 11, s = row & (SEQ - 1);
          const int h = col >> 6, d = col & (HDIM - 1);
          out[(size_t)(((bb * NHEAD + h) * SEQ) + s) * HDIM + d] = (OT)val;
        } else {
          out[(size_t)row * N + col] = (OT)val;
        }
      }
}

// ---------------------------------------------------------------------------
// Flash-attention (causal, online softmax). Q/K/V in [B,H,S,D] f16.
// One block per (64-query tile, batch*head). O accumulators in registers.
// K/V tiles register-prefetched one iteration ahead; V staged transposed
// (VsT[d][key]) so P*V B-fragments are vector LDS loads.
// ---------------------------------------------------------------------------
__global__ __launch_bounds__(256) void attn_flash(
    const _Float16* __restrict__ Qh, const _Float16* __restrict__ Kh,
    const _Float16* __restrict__ Vh, _Float16* __restrict__ Ah) {
  constexpr int TQ = 64, TK = 64;
  constexpr int LDHP = HDIM + 8;  // 72: f16 arrays, 16B-aligned rows
  constexpr int LDS_S = 68;       // f32 score array stride
  __shared__ __align__(16) _Float16 Qs[TQ * LDHP];    // [q][d]
  __shared__ __align__(16) _Float16 Ks[TK * LDHP];    // [key][d]  (K-major for Q*K^T)
  __shared__ __align__(16) _Float16 VsT[HDIM * LDHP]; // [d][key]  (K-major for P*V)
  __shared__ __align__(16) float    Ss[TQ * LDS_S];
  __shared__ __align__(16) _Float16 Ps[TQ * LDHP];    // [q][key]
  __shared__ float row_m[TQ], row_l[TQ], row_scale[TQ];

  const int tid  = threadIdx.x;
  const int lane = tid & 31;
  const int wave = tid >> 5;
  const int tr   = wave >> 1;  // 16-row tile index 0..3
  const int tc0  = wave & 1;   // handles col tiles tc0 and tc0+2
  const int q0   = blockIdx.x * TQ;
  const int bh   = blockIdx.y;  // b*NHEAD + h
  const int b    = bh >> 4, h = bh & 15;
  const size_t head_off = (size_t)bh * SEQ * HDIM;

#pragma unroll
  for (int i = 0; i < 2; ++i) {
    const int idx = tid + i * 256;  // 512 v8h = 64x64
    const int r = idx >> 3, d = (idx & 7) * 8;
    *(v8h*)&Qs[r * LDHP + d] =
        *(const v8h*)&Qh[head_off + (size_t)(q0 + r) * HDIM + d];
  }
  if (tid < TQ) { row_m[tid] = -3.0e38f; row_l[tid] = 0.0f; }

  v8h kReg[2], vReg[2];
  auto load_kv = [&](int j0) {
#pragma unroll
    for (int i = 0; i < 2; ++i) {
      const int idx = tid + i * 256;
      const int r = idx >> 3, d = (idx & 7) * 8;
      kReg[i] = *(const v8h*)&Kh[head_off + (size_t)(j0 + r) * HDIM + d];
      vReg[i] = *(const v8h*)&Vh[head_off + (size_t)(j0 + r) * HDIM + d];
    }
  };
  load_kv(0);

  v8f o[2] = {};
  const int cn = lane & 15, g = lane >> 4;

  for (int j0 = 0; j0 <= q0; j0 += TK) {
    __syncthreads();  // previous iteration done with Ks/VsT/Ps
#pragma unroll
    for (int i = 0; i < 2; ++i) {
      const int idx = tid + i * 256;
      const int r = idx >> 3, d = (idx & 7) * 8;
      *(v8h*)&Ks[r * LDHP + d] = kReg[i];
#pragma unroll
      for (int t = 0; t < 8; ++t) VsT[(d + t) * LDHP + r] = vReg[i][t];
    }
    __syncthreads();
    if (j0 + TK <= q0) load_kv(j0 + TK);  // overlap with compute below

    // S = Q * K^T  (1/sqrt(D) folded into Q at projection time)
#pragma unroll
    for (int t = 0; t < 2; ++t) {
      const int tc = tc0 + 2 * t;
      v8f s = {};
#pragma unroll
      for (int kc = 0; kc < 2; ++kc) {
        const v16h aq = frag_a(&Qs[(tr * 16) * LDHP + kc * 32], LDHP, lane);
        const v16h bk = frag_b_kmaj(&Ks[(tc * 16) * LDHP + kc * 32], LDHP, lane);
        s = __builtin_amdgcn_wmma_f32_16x16x32_f16(false, aq, false, bk,
                                                   (short)0, s, false, false);
      }
#pragma unroll
      for (int v = 0; v < 8; ++v)
        Ss[(tr * 16 + g * 8 + v) * LDS_S + tc * 16 + cn] = s[v];
    }
    __syncthreads();

    // Online softmax: one thread per query row.
    if (tid < TQ) {
      const int r = tid;
      const int limit = (j0 == q0) ? r : (TK - 1);  // causal diagonal tile
      const float m_old = row_m[r];
      float mval = m_old;
      for (int c = 0; c <= limit; ++c) mval = fmaxf(mval, Ss[r * LDS_S + c]);
      const float alpha = __expf(m_old - mval);
      float sum = 0.0f;
      for (int c0 = 0; c0 < TK; c0 += 8) {
        v8h pv;
#pragma unroll
        for (int u = 0; u < 8; ++u) {
          const int c = c0 + u;
          const float p =
              (c <= limit) ? __expf(Ss[r * LDS_S + c] - mval) : 0.0f;
          pv[u] = (_Float16)p;
          sum += p;
        }
        *(v8h*)&Ps[r * LDHP + c0] = pv;
      }
      row_m[r] = mval;
      row_l[r] = row_l[r] * alpha + sum;
      row_scale[r] = alpha;
    }
    __syncthreads();

    // Rescale running O and accumulate P * V
#pragma unroll
    for (int t = 0; t < 2; ++t) {
#pragma unroll
      for (int v = 0; v < 8; ++v)
        o[t][v] *= row_scale[tr * 16 + g * 8 + v];
      const int tc = tc0 + 2 * t;
#pragma unroll
      for (int kc = 0; kc < 2; ++kc) {
        const v16h ap = frag_a(&Ps[(tr * 16) * LDHP + kc * 32], LDHP, lane);
        const v16h bv =
            frag_b_kmaj(&VsT[(tc * 16) * LDHP + kc * 32], LDHP, lane);
        o[t] = __builtin_amdgcn_wmma_f32_16x16x32_f16(false, ap, false, bv,
                                                      (short)0, o[t], false, false);
      }
    }
  }

  // Normalize and write attention output as [m = b*S+s, e = h*D+d] f16
#pragma unroll
  for (int t = 0; t < 2; ++t) {
    const int tc = tc0 + 2 * t;
#pragma unroll
    for (int v = 0; v < 8; ++v) {
      const int row = tr * 16 + g * 8 + v;
      const int d   = tc * 16 + cn;
      const float val = o[t][v] / row_l[row];
      Ah[(size_t)(b * SEQ + q0 + row) * EMBED + h * HDIM + d] = (_Float16)val;
    }
  }
}

// ---------------------------------------------------------------------------
extern "C" void kernel_launch(void* const* d_in, const int* in_sizes, int n_in,
                              void* d_out, int out_size, void* d_ws, size_t ws_size,
                              hipStream_t stream) {
  (void)in_sizes; (void)n_in; (void)out_size; (void)ws_size;
  const float* x  = (const float*)d_in[0];
  // d_in[1] = causal mask (bool) -- handled analytically, not read
  const float* Wq = (const float*)d_in[2];
  const float* bq = (const float*)d_in[3];
  const float* Wk = (const float*)d_in[4];
  const float* bk = (const float*)d_in[5];
  const float* Wv = (const float*)d_in[6];
  const float* bv = (const float*)d_in[7];
  const float* Wo = (const float*)d_in[8];
  const float* bo = (const float*)d_in[9];
  float* out = (float*)d_out;

  const size_t head_elems = (size_t)BATCH * NHEAD * SEQ * HDIM;  // 8M f16
  _Float16* Qh = (_Float16*)d_ws;
  _Float16* Kh = Qh + head_elems;
  _Float16* Vh = Kh + head_elems;
  _Float16* Ah = Vh + head_elems;  // [MTOT, EMBED] f16

  const dim3 gproj(EMBED / 128, MTOT / 128);  // (8, 64)
  const dim3 blk(256);
  const float inv_sqrt_d = 0.125f;  // 1/sqrt(64), folded into Q

  gemm_wmma<float, _Float16, true><<<gproj, blk, 0, stream>>>(
      x, Wq, bq, Qh, MTOT, EMBED, EMBED, inv_sqrt_d);
  gemm_wmma<float, _Float16, true><<<gproj, blk, 0, stream>>>(
      x, Wk, bk, Kh, MTOT, EMBED, EMBED, 1.0f);
  gemm_wmma<float, _Float16, true><<<gproj, blk, 0, stream>>>(
      x, Wv, bv, Vh, MTOT, EMBED, EMBED, 1.0f);

  const dim3 gattn(SEQ / 64, BATCH * NHEAD);  // (32, 64)
  attn_flash<<<gattn, blk, 0, stream>>>(Qh, Kh, Vh, Ah);

  gemm_wmma<_Float16, float, false><<<gproj, blk, 0, stream>>>(
      Ah, Wo, bo, out, MTOT, EMBED, EMBED, 1.0f);
}
